// KNNEdgeBuilder_24970939859602
// MI455X (gfx1250) — matrix operand order repaired and checked by reading, not compiled, not run on hardware
//
#include <hip/hip_runtime.h>
#include <hip/hip_bf16.h>
#include <math.h>

// Problem constants (from the reference): B=8, N=4096, C=256, k=4.
#define NB 8
#define NN 4096
#define NC 256
#define KT 4
#define NTILES (NN / 16)   // 256 column tiles of width 16
#define WAVES  4           // waves per workgroup in the kNN kernel

typedef __attribute__((ext_vector_type(16))) _Float16 v16h;
typedef __attribute__((ext_vector_type(8)))  _Float16 h8;
typedef __attribute__((ext_vector_type(8)))  float    v8f;
typedef __attribute__((ext_vector_type(4)))  float    f4;

// ---------------------------------------------------------------------------
// Kernel 1: L2-normalize each 256-float row and emit f16 into workspace.
// One wave32 per row, 8 floats per lane, butterfly reduce for the norm.
// ---------------------------------------------------------------------------
__global__ __launch_bounds__(256)
void normalize_rows_f16(const float* __restrict__ x, _Float16* __restrict__ y) {
  const int wave = threadIdx.x >> 5;
  const int lane = threadIdx.x & 31;
  const size_t row = (size_t)blockIdx.x * 8 + wave;

  const float* p = x + row * NC + (size_t)lane * 8;
  f4 a  = *(const f4*)p;
  f4 b2 = *(const f4*)(p + 4);

  float s = 0.f;
#pragma unroll
  for (int i = 0; i < 4; ++i) s += a[i] * a[i] + b2[i] * b2[i];
#pragma unroll
  for (int m = 16; m >= 1; m >>= 1) s += __shfl_xor(s, m, 32);

  const float r = rsqrtf(s);
  h8 o;
#pragma unroll
  for (int i = 0; i < 4; ++i) {
    o[i]     = (_Float16)(a[i]  * r);
    o[4 + i] = (_Float16)(b2[i] * r);
  }
  *(h8*)(y + row * NC + (size_t)lane * 8) = o;
}

__device__ __forceinline__ uint32_t umax32(uint32_t a, uint32_t b) { return a > b ? a : b; }
__device__ __forceinline__ uint32_t umin32(uint32_t a, uint32_t b) { return a < b ? a : b; }

// (value, column) packed into one monotonic 32-bit key:
//   top 20 bits: sign-rectified float bits (unsigned order == float order)
//   low 12 bits: 0xFFF - col  (ties resolved toward the LOWER column index)
__device__ __forceinline__ uint32_t make_key(float v, int col) {
  uint32_t u = __float_as_uint(v);
  u ^= (uint32_t)((int32_t)u >> 31) | 0x80000000u;
  return (u & 0xFFFFF000u) | (uint32_t)(0xFFF - col);
}

// Select-free sorted top-4 insertion: 7 u32 min/max ops, no cndmask chains.
__device__ __forceinline__ void top4_insert_key(uint32_t (&k)[4], uint32_t nk) {
  uint32_t t;
  t = umin32(k[0], nk); k[0] = umax32(k[0], nk); nk = t;
  t = umin32(k[1], nk); k[1] = umax32(k[1], nk); nk = t;
  t = umin32(k[2], nk); k[2] = umax32(k[2], nk); nk = t;
  k[3] = umax32(k[3], nk);
}

// ---------------------------------------------------------------------------
// Kernel 2: fused sim-tile WMMA + per-row top-4 on packed keys.
// B panel double-buffered across column tiles so L2 latency hides behind the
// previous tile's WMMA + selection work.
// ---------------------------------------------------------------------------
__global__ __launch_bounds__(128)
void knn_topk(const _Float16* __restrict__ fn,
              int* __restrict__ edge_index, float* __restrict__ edge_weight) {
  const int b    = blockIdx.x / NTILES;
  const int it   = blockIdx.x % NTILES;
  const int row0 = it * 16;

  const int tid  = threadIdx.x;
  const int wave = tid >> 5;
  const int lane = tid & 31;
  const int half = lane >> 4;   // 0: K-low / rows 0-7,  1: K-high / rows 8-15
  const int nl   = lane & 15;   // A: row-in-tile, B/C: col-in-tile

  const _Float16* base = fn + (size_t)b * NN * NC;

  // ---- A fragments: rows [row0, row0+16), K = 256 as 8 chunks of 32.
  // Layout (16-bit A 16x32): lane holds M = lane%16; lane-half selects
  // K run [h*16, h*16+16) (matches B)  -> two contiguous 16B loads.
  v16h afrag[8];
  const _Float16* arow = base + (size_t)(row0 + nl) * NC + half * 16;
#pragma unroll
  for (int kc = 0; kc < 8; ++kc) {
    h8 a0 = *(const h8*)(arow + kc * 32);
    h8 a1 = *(const h8*)(arow + kc * 32 + 8);
#pragma unroll
    for (int i = 0; i < 8; ++i) { afrag[kc][i] = a0[i]; afrag[kc][8 + i] = a1[i]; }
  }

  // Per-lane top-4 key lists for the 8 rows this lane-half produces.
  uint32_t keys[8][4];
#pragma unroll
  for (int r = 0; r < 8; ++r)
#pragma unroll
    for (int e = 0; e < 4; ++e) keys[r][e] = 0u;   // below any real key

  // B fragment address for tile jt: lane holds N = jt*16 + lane%16, with
  // K in [half*16, half*16+16) per 32-chunk -> one contiguous 32B run.
  const _Float16* bbase = base + (size_t)nl * NC + half * 16;
#define BROW(jt) (bbase + (size_t)(jt) * (16 * NC))

  // ---- Ping-pong B panels; manual 2x unroll keeps register indices static.
  v16h b0[8], b1[8];
#pragma unroll
  for (int kc = 0; kc < 8; ++kc) b0[kc] = *(const v16h*)(BROW(wave) + kc * 32);

  for (int jt = wave; jt < NTILES; jt += 2 * WAVES) {
    const int jt1 = jt + WAVES;                                   // < NTILES
    const int jt2 = (jt1 + WAVES < NTILES) ? jt1 + WAVES : wave;  // wrapped

    // Issue loads for tile jt1 while computing tile jt.
#pragma unroll
    for (int kc = 0; kc < 8; ++kc) b1[kc] = *(const v16h*)(BROW(jt1) + kc * 32);

    {
      const int ncol = jt * 16 + nl;
      v8f c0 = {}, c1 = {};
#pragma unroll
      for (int kc = 0; kc < 8; kc += 2) {
        c0 = __builtin_amdgcn_wmma_f32_16x16x32_f16(false, afrag[kc],     false,
                                                    b0[kc],     (short)0, c0, false, false);
        c1 = __builtin_amdgcn_wmma_f32_16x16x32_f16(false, afrag[kc + 1], false,
                                                    b0[kc + 1], (short)0, c1, false, false);
      }
#pragma unroll
      for (int r = 0; r < 8; ++r) {
        const int mg = row0 + half * 8 + r;
        uint32_t nk = make_key(c0[r] + c1[r], ncol);
        nk = (ncol == mg) ? 0u : nk;                 // mask self-loop
        top4_insert_key(keys[r], nk);
      }
    }

    // Issue loads for tile jt2 while computing tile jt1.
#pragma unroll
    for (int kc = 0; kc < 8; ++kc) b0[kc] = *(const v16h*)(BROW(jt2) + kc * 32);

    {
      const int ncol = jt1 * 16 + nl;
      v8f c0 = {}, c1 = {};
#pragma unroll
      for (int kc = 0; kc < 8; kc += 2) {
        c0 = __builtin_amdgcn_wmma_f32_16x16x32_f16(false, afrag[kc],     false,
                                                    b1[kc],     (short)0, c0, false, false);
        c1 = __builtin_amdgcn_wmma_f32_16x16x32_f16(false, afrag[kc + 1], false,
                                                    b1[kc + 1], (short)0, c1, false, false);
      }
#pragma unroll
      for (int r = 0; r < 8; ++r) {
        const int mg = row0 + half * 8 + r;
        uint32_t nk = make_key(c0[r] + c1[r], ncol);
        nk = (ncol == mg) ? 0u : nk;
        top4_insert_key(keys[r], nk);
      }
    }
  }
#undef BROW

  // ---- Merge per-lane lists through LDS (exact: the global top-4 is
  // contained in the union of per-sublist top-4s; keys are totally ordered,
  // so the result is deterministic regardless of merge order).
  // Layout [slot][row][e] with +1-row pad: 32-lane writes (slot stride 68
  // dwords -> bank step 4) and 16-thread reads (row stride 4) conflict-free.
  __shared__ uint32_t s_key[WAVES * 16][16 + 1][4];   // ~17 KB
  const int slot = wave * 16 + nl;
#pragma unroll
  for (int r = 0; r < 8; ++r)
#pragma unroll
    for (int e = 0; e < 4; ++e) s_key[slot][half * 8 + r][e] = keys[r][e];
  __syncthreads();

  if (tid < 16) {
    const int row = tid;
    uint32_t bk[4] = {0u, 0u, 0u, 0u};
    for (int s2 = 0; s2 < WAVES * 16; ++s2)
#pragma unroll
      for (int e = 0; e < 4; ++e) top4_insert_key(bk, s_key[s2][row][e]);

    const int g   = row0 + row;
    const int NKE = NN * KT;                       // edges per batch
    int*   src = edge_index + (size_t)b * 2 * NKE; // (B, 2, N*k): src row
    int*   tgt = src + NKE;                        //              tgt row
    float* w   = edge_weight + (size_t)b * NKE;

    const _Float16* ra = base + (size_t)g * NC;
#pragma unroll
    for (int e = 0; e < 4; ++e) {
      const int col = 0xFFF - (int)(bk[e] & 0xFFFu);
      // Exact edge weight: f32 dot of the f16-normalized rows.
      const _Float16* rb = base + (size_t)col * NC;
      float sdot = 0.f;
      for (int i = 0; i < NC; i += 8) {
        h8 xa = *(const h8*)(ra + i);
        h8 xb = *(const h8*)(rb + i);
#pragma unroll
        for (int j = 0; j < 8; ++j) sdot += (float)xa[j] * (float)xb[j];
      }
      src[g * 4 + e] = g;
      tgt[g * 4 + e] = col;
      w[g * 4 + e]   = sdot;
    }
  }
}

extern "C" void kernel_launch(void* const* d_in, const int* in_sizes, int n_in,
                              void* d_out, int out_size, void* d_ws, size_t ws_size,
                              hipStream_t stream) {
  const float* x = (const float*)d_in[0];          // (8, 4096, 256) fp32

  // Workspace: normalized f16 features, 8*4096*256*2 = 16 MB (< ws_size).
  _Float16* fnh = (_Float16*)d_ws;

  // d_out: edge_index (8,2,16384) int32 then edge_weight (8,16384) f32.
  int*   edge_index  = (int*)d_out;
  float* edge_weight = (float*)d_out + (size_t)NB * 2 * NN * KT;

  // 32768 rows, one wave each, 8 waves per 256-thread block.
  normalize_rows_f16<<<(NB * NN) / 8, 256, 0, stream>>>(x, fnh);

  // One block per (batch, 16-row tile): 8 * 256 = 2048 blocks of 128 threads.
  knn_topk<<<NB * NTILES, 128, 0, stream>>>(fnh, edge_index, edge_weight);
}
